// TransformerLayerQKV_82291573392117
// MI455X (gfx1250) — matrix operand order repaired and checked by reading, not compiled, not run on hardware
//
#include <hip/hip_runtime.h>
#include <hip/hip_bf16.h>

typedef _Float16 v16h __attribute__((ext_vector_type(16)));
typedef _Float16 v8h  __attribute__((ext_vector_type(8)));
typedef float    v8f  __attribute__((ext_vector_type(8)));

// ---------------------------------------------------------------------------
// CDNA5 async global->LDS copy (ASYNCcnt-tracked), per-lane 16B
// ---------------------------------------------------------------------------
__device__ __forceinline__ void async_copy_b128(unsigned lds_addr, const void* gptr) {
  asm volatile("global_load_async_to_lds_b128 %0, %1, off"
               :: "v"(lds_addr), "v"(gptr) : "memory");
}
__device__ __forceinline__ void wait_async_all() {
  asm volatile("s_wait_asynccnt 0x0" ::: "memory");
}
// Generic shared pointer -> wave-relative LDS byte address (aperture low bits)
__device__ __forceinline__ unsigned lds_lo(const void* p) {
  return (unsigned)(size_t)p;
}

// ---------------------------------------------------------------------------
// WMMA helpers (CDNA5 wave32, 16x16x32 f16 -> f32 accumulate)
// ---------------------------------------------------------------------------
__device__ __forceinline__ v8f wmma_f16(v16h a, v16h b, v8f c) {
  return __builtin_amdgcn_wmma_f32_16x16x32_f16(
      /*neg_a=*/false, a, /*neg_b=*/false, b,
      /*c_mod=*/(short)0, c, /*reuse_a=*/false, /*reuse_b=*/false);
}

__device__ __forceinline__ v16h make_v16(v8h x, v8h y) {
  return __builtin_shufflevector(x, y, 0, 1, 2, 3, 4, 5, 6, 7,
                                 8, 9, 10, 11, 12, 13, 14, 15);
}

// A operand fragment (16x32 f16). Lane l<16: row=row0+l, K = kbase+{0..7,16..23}
// Lane l>=16: row=row0+l-16, K = kbase+{8..15,24..31}. tile row-major [rows][ld].
__device__ __forceinline__ v16h frag_a(const _Float16* tile, int ld, int row0, int kbase) {
  const int lane = threadIdx.x & 31;
  const int lo = lane & 15;
  const int hk = (lane >= 16) ? 8 : 0;
  const _Float16* p = tile + (size_t)(row0 + lo) * ld + kbase + hk;
  v8h x = *(const v8h*)p;
  v8h y = *(const v8h*)(p + 16);
  return make_v16(x, y);
}

// B operand fragment (32x16 f16) from an N-major tile (tile[n][k]).
__device__ __forceinline__ v16h frag_b(const _Float16* tile, int ld, int col0, int kbase) {
  const int lane = threadIdx.x & 31;
  const int lo = lane & 15;
  const int hk = (lane >= 16) ? 8 : 0;
  const _Float16* p = tile + (size_t)(col0 + lo) * ld + kbase + hk;
  v8h x = *(const v8h*)p;
  v8h y = *(const v8h*)(p + 16);
  return make_v16(x, y);
}

// ---------------------------------------------------------------------------
// LayerNorm fp32 -> fp16, one block per row
// ---------------------------------------------------------------------------
__global__ __launch_bounds__(256) void layernorm_f16_kernel(
    const float* __restrict__ x, const float* __restrict__ w,
    const float* __restrict__ b, _Float16* __restrict__ out, int C) {
  __shared__ float sbuf[16];
  const int row = blockIdx.x;
  const int tid = threadIdx.x;
  const float* xr = x + (size_t)row * C;
  float s = 0.f, s2 = 0.f;
  for (int i = tid; i < C; i += 256) {
    float v = xr[i];
    s += v;
    s2 += v * v;
  }
#pragma unroll
  for (int sh = 16; sh >= 1; sh >>= 1) {
    s += __shfl_xor(s, sh, 32);
    s2 += __shfl_xor(s2, sh, 32);
  }
  const int wid = tid >> 5, lane = tid & 31;
  if (lane == 0) {
    sbuf[wid] = s;
    sbuf[8 + wid] = s2;
  }
  __syncthreads();
  float ts = 0.f, ts2 = 0.f;
#pragma unroll
  for (int i = 0; i < 8; ++i) {
    ts += sbuf[i];
    ts2 += sbuf[8 + i];
  }
  const float mu = ts / (float)C;
  const float var = ts2 / (float)C - mu * mu;
  const float rstd = rsqrtf(var + 1e-5f);
  for (int i = tid; i < C; i += 256)
    out[(size_t)row * C + i] = (_Float16)((xr[i] - mu) * rstd * w[i] + b[i]);
}

// ---------------------------------------------------------------------------
// fp32 -> fp16 flat convert
// ---------------------------------------------------------------------------
__global__ void cvt_f16_kernel(const float* __restrict__ src,
                               _Float16* __restrict__ dst, size_t n) {
  for (size_t i = (size_t)blockIdx.x * blockDim.x + threadIdx.x; i < n;
       i += (size_t)gridDim.x * blockDim.x)
    dst[i] = (_Float16)src[i];
}

// fp32 KxN (row major) -> fp16 NxK (row major)
__global__ void cvt_transpose_kernel(const float* __restrict__ src,
                                     _Float16* __restrict__ dst, int K, int N) {
  const size_t total = (size_t)K * N;
  for (size_t i = (size_t)blockIdx.x * blockDim.x + threadIdx.x; i < total;
       i += (size_t)gridDim.x * blockDim.x) {
    const int k = (int)(i / N);
    const int n = (int)(i % N);
    dst[(size_t)n * K + k] = (_Float16)src[i];
  }
}

// V (B,M,H,D) f16 -> Vt (B,H,D,M) f16
__global__ void transpose_v_kernel(const _Float16* __restrict__ V,
                                   _Float16* __restrict__ Vt, int B, int M,
                                   int H, int D) {
  const size_t total = (size_t)B * M * H * D;
  for (size_t i = (size_t)blockIdx.x * blockDim.x + threadIdx.x; i < total;
       i += (size_t)gridDim.x * blockDim.x) {
    const int d = (int)(i % D);
    size_t t = i / D;
    const int h = (int)(t % H);
    t /= H;
    const int m = (int)(t % M);
    const int b = (int)(t / M);
    Vt[(((size_t)b * H + h) * D + d) * M + m] = V[i];
  }
}

// ---------------------------------------------------------------------------
// WMMA GEMM: out[M,N] = A[M,K] f16 * Bt[N,K]^T f16 (+bias,+ReLU,+f32 resid).
// 256 threads = 8 waves. Tile 128x128x32, double-buffered async LDS staging.
// ---------------------------------------------------------------------------
__global__ __launch_bounds__(256) void wmma_gemm_kernel(
    const _Float16* __restrict__ A, const _Float16* __restrict__ Bt,
    const float* __restrict__ bias, const float* __restrict__ resid,
    float* __restrict__ outF, _Float16* __restrict__ outH, int Mtot, int Ntot,
    int Ktot, int relu) {
  __shared__ _Float16 As[2][128][48];
  __shared__ _Float16 Bs[2][128][48];

  const int tid = threadIdx.x;
  const int lane = tid & 31;
  const int wid = tid >> 5;
  const int wm = wid & 3;   // 32-row group
  const int wn = wid >> 2;  // 64-col group
  const int m0 = blockIdx.y * 128;
  const int n0 = blockIdx.x * 128;

  // Per-thread staging coordinates: rows (tid>>2) and (tid>>2)+64, 16B chunk.
  const int sr = tid >> 2;          // 0..63
  const int so = (tid & 3) * 8;     // half offset within 32-wide K slab
  const _Float16* Ag0 = A + (size_t)(m0 + sr) * Ktot + so;
  const _Float16* Ag1 = Ag0 + (size_t)64 * Ktot;
  const _Float16* Bg0 = Bt + (size_t)(n0 + sr) * Ktot + so;
  const _Float16* Bg1 = Bg0 + (size_t)64 * Ktot;
  unsigned lA0[2], lA1[2], lB0[2], lB1[2];
#pragma unroll
  for (int s = 0; s < 2; ++s) {
    lA0[s] = lds_lo(&As[s][sr][so]);
    lA1[s] = lds_lo(&As[s][sr + 64][so]);
    lB0[s] = lds_lo(&Bs[s][sr][so]);
    lB1[s] = lds_lo(&Bs[s][sr + 64][so]);
  }

  v8f acc[2][4] = {};

  // Prologue: stage first K-slab into buffer 0.
  async_copy_b128(lA0[0], Ag0);
  async_copy_b128(lA1[0], Ag1);
  async_copy_b128(lB0[0], Bg0);
  async_copy_b128(lB1[0], Bg1);

  int buf = 0;
  for (int k0 = 0; k0 < Ktot; k0 += 32) {
    wait_async_all();      // current slab landed in LDS (per-wave)
    __syncthreads();       // everyone landed; prev reads of other buf done
    if (k0 + 32 < Ktot) {  // overlap next slab with compute
      const int nb = buf ^ 1;
      async_copy_b128(lA0[nb], Ag0 + k0 + 32);
      async_copy_b128(lA1[nb], Ag1 + k0 + 32);
      async_copy_b128(lB0[nb], Bg0 + k0 + 32);
      async_copy_b128(lB1[nb], Bg1 + k0 + 32);
    }

    // Batch all fragment loads into distinct registers, then burst 8 WMMAs so
    // the ds_loads can be issued together and drained with partial waits.
    const _Float16* at = &As[buf][0][0];
    const _Float16* bt = &Bs[buf][0][0];
    v16h a0 = frag_a(at, 48, wm * 32, 0);
    v16h a1 = frag_a(at, 48, wm * 32 + 16, 0);
    v16h bf[4];
#pragma unroll
    for (int nt = 0; nt < 4; ++nt) bf[nt] = frag_b(bt, 48, wn * 64 + nt * 16, 0);
#pragma unroll
    for (int nt = 0; nt < 4; ++nt) {
      acc[0][nt] = wmma_f16(a0, bf[nt], acc[0][nt]);
      acc[1][nt] = wmma_f16(a1, bf[nt], acc[1][nt]);
    }
    buf ^= 1;
  }

  // Epilogue. C/D layout: lane<16 -> N=lane, rows v ; lane>=16 -> rows v+8.
  const int lc = lane & 15;
  const int rb = (lane < 16) ? 0 : 8;
#pragma unroll
  for (int mt = 0; mt < 2; ++mt) {
#pragma unroll
    for (int nt = 0; nt < 4; ++nt) {
      const int col = n0 + wn * 64 + nt * 16 + lc;
      const float bv = bias ? bias[col] : 0.f;
#pragma unroll
      for (int v = 0; v < 8; ++v) {
        const int row = m0 + wm * 32 + mt * 16 + rb + v;
        float val = acc[mt][nt][v] + bv;
        if (relu) val = fmaxf(val, 0.f);
        if (resid) val += resid[(size_t)row * Ntot + col];
        if (outF) outF[(size_t)row * Ntot + col] = val;
        if (outH) outH[(size_t)row * Ntot + col] = (_Float16)val;
      }
    }
  }
}

// ---------------------------------------------------------------------------
// Flash attention: block = (b, h, 64 query rows), 128 threads = 4 waves.
// D = 64. Online softmax; S and P·V via WMMA; async double-buffered K/V tiles.
// ---------------------------------------------------------------------------
__global__ __launch_bounds__(128) void flash_attn_kernel(
    const _Float16* __restrict__ Q, const _Float16* __restrict__ K,
    const _Float16* __restrict__ Vt, _Float16* __restrict__ O, int B, int N,
    int Mseq, int H, int D) {
  __shared__ _Float16 Qs[64][72];
  __shared__ _Float16 Ks[2][64][72];
  __shared__ _Float16 Vs[2][64][72];  // Vt tile: [d][m]
  __shared__ _Float16 Ps[64][72];

  const int tid = threadIdx.x;
  const int lane = tid & 31;
  const int wid = tid >> 5;
  const int b = blockIdx.z;
  const int h = blockIdx.y;
  const int n0 = blockIdx.x * 64;
  const int C = H * D;
  const int qr = wid * 16;
  const int lc = lane & 15;
  const int rb = (lane < 16) ? 0 : 8;
  const float scale = rsqrtf((float)D);

  // Per-thread staging coordinates: 4 row-groups of 16, 16B chunk each.
  const int sr = tid >> 3;        // 0..15
  const int so = (tid & 7) * 8;   // 0..56

  // Stage Q tile (64x64) once, asynchronously.
  {
    const _Float16* Qg = Q + ((size_t)(b * N + n0 + sr)) * C + h * D + so;
#pragma unroll
    for (int g = 0; g < 4; ++g)
      async_copy_b128(lds_lo(&Qs[sr + 16 * g][so]), Qg + (size_t)(16 * g) * C);
  }
  // Stage first K/V chunk into buffer 0.
  {
    const _Float16* Kg = K + ((size_t)(b * Mseq + sr)) * C + h * D + so;
    const _Float16* Vg = Vt + (((size_t)b * H + h) * D + sr) * Mseq + so;
#pragma unroll
    for (int g = 0; g < 4; ++g) {
      async_copy_b128(lds_lo(&Ks[0][sr + 16 * g][so]), Kg + (size_t)(16 * g) * C);
      async_copy_b128(lds_lo(&Vs[0][sr + 16 * g][so]), Vg + (size_t)(16 * g) * Mseq);
    }
  }

  v8f o_acc[4] = {};
  float mrow[8], lrow[8];
#pragma unroll
  for (int v = 0; v < 8; ++v) {
    mrow[v] = -1e30f;
    lrow[v] = 0.f;
  }

  int buf = 0;
  for (int m0 = 0; m0 < Mseq; m0 += 64) {
    wait_async_all();
    __syncthreads();
    if (m0 + 64 < Mseq) {  // prefetch next K/V chunk into other buffer
      const int nb = buf ^ 1;
      const _Float16* Kg =
          K + ((size_t)(b * Mseq + m0 + 64 + sr)) * C + h * D + so;
      const _Float16* Vg =
          Vt + (((size_t)b * H + h) * D + sr) * Mseq + m0 + 64 + so;
#pragma unroll
      for (int g = 0; g < 4; ++g) {
        async_copy_b128(lds_lo(&Ks[nb][sr + 16 * g][so]),
                        Kg + (size_t)(16 * g) * C);
        async_copy_b128(lds_lo(&Vs[nb][sr + 16 * g][so]),
                        Vg + (size_t)(16 * g) * Mseq);
      }
    }

    // S = Q * K^T  (16 rows x 64 keys per wave); batch frags, burst WMMAs.
    v8f s[4] = {};
#pragma unroll
    for (int kk = 0; kk < 64; kk += 32) {
      v16h aq = frag_a(&Qs[0][0], 72, qr, kk);
      v16h bk[4];
#pragma unroll
      for (int mt = 0; mt < 4; ++mt)
        bk[mt] = frag_b(&Ks[buf][0][0], 72, mt * 16, kk);
#pragma unroll
      for (int mt = 0; mt < 4; ++mt) s[mt] = wmma_f16(aq, bk[mt], s[mt]);
    }

    // Online softmax (row stats reduced over the 16 lanes of each half-wave).
#pragma unroll
    for (int v = 0; v < 8; ++v) {
      float mx = -1e30f;
#pragma unroll
      for (int mt = 0; mt < 4; ++mt) {
        s[mt][v] *= scale;
        mx = fmaxf(mx, s[mt][v]);
      }
#pragma unroll
      for (int sh = 1; sh < 16; sh <<= 1) mx = fmaxf(mx, __shfl_xor(mx, sh, 16));
      const float mnew = fmaxf(mrow[v], mx);
      const float alpha = __expf(mrow[v] - mnew);
      float rs = 0.f;
#pragma unroll
      for (int mt = 0; mt < 4; ++mt) {
        const float p = __expf(s[mt][v] - mnew);
        s[mt][v] = p;
        rs += p;
      }
#pragma unroll
      for (int sh = 1; sh < 16; sh <<= 1) rs += __shfl_xor(rs, sh, 16);
      lrow[v] = lrow[v] * alpha + rs;
      mrow[v] = mnew;
#pragma unroll
      for (int dt = 0; dt < 4; ++dt) o_acc[dt][v] *= alpha;
#pragma unroll
      for (int mt = 0; mt < 4; ++mt)
        Ps[qr + rb + v][mt * 16 + lc] = (_Float16)s[mt][v];
    }
    __syncthreads();  // make P visible for A-operand fragment loads

    // O += P * V   (B operand = Vt tile, d-major); batch frags, burst WMMAs.
#pragma unroll
    for (int kk = 0; kk < 64; kk += 32) {
      v16h ap = frag_a(&Ps[0][0], 72, qr, kk);
      v16h bv[4];
#pragma unroll
      for (int dt = 0; dt < 4; ++dt)
        bv[dt] = frag_b(&Vs[buf][0][0], 72, dt * 16, kk);
#pragma unroll
      for (int dt = 0; dt < 4; ++dt) o_acc[dt] = wmma_f16(ap, bv[dt], o_acc[dt]);
    }
    buf ^= 1;
  }

  // Normalize and write (B,N,C) f16
#pragma unroll
  for (int v = 0; v < 8; ++v) {
    const float inv = 1.f / lrow[v];
    const int row = n0 + qr + rb + v;
#pragma unroll
    for (int dt = 0; dt < 4; ++dt)
      O[((size_t)(b * N + row)) * C + h * D + dt * 16 + lc] =
          (_Float16)(o_acc[dt][v] * inv);
  }
}

// ---------------------------------------------------------------------------
// Host-side orchestration
// ---------------------------------------------------------------------------
extern "C" void kernel_launch(void* const* d_in, const int* in_sizes, int n_in,
                              void* d_out, int out_size, void* d_ws,
                              size_t ws_size, hipStream_t stream) {
  (void)in_sizes; (void)n_in; (void)out_size; (void)ws_size;
  const float* q = (const float*)d_in[0];
  const float* k = (const float*)d_in[1];
  const float* v = (const float*)d_in[2];
  const float* Wq = (const float*)d_in[3];
  const float* Wk = (const float*)d_in[4];
  const float* Wv = (const float*)d_in[5];
  const float* Wp = (const float*)d_in[6];
  const float* bp = (const float*)d_in[7];
  const float* ln1_w = (const float*)d_in[8];
  const float* ln1_b = (const float*)d_in[9];
  const float* ln2_w = (const float*)d_in[10];
  const float* ln2_b = (const float*)d_in[11];
  const float* W1 = (const float*)d_in[12];
  const float* b1 = (const float*)d_in[13];
  const float* W2 = (const float*)d_in[14];
  const float* b2 = (const float*)d_in[15];
  float* out = (float*)d_out;

  const int B = 4, N = 2048, Mseq = 2048, C = 768, H = 12, D = 64, DFF = 3072;
  const int BN = B * N;

  char* ws = (char*)d_ws;
  size_t off = 0;
  auto alloc = [&](size_t bytes) -> void* {
    void* p = (void*)(ws + off);
    off += (bytes + 255) & ~(size_t)255;
    return p;
  };

  _Float16* qn16 = (_Float16*)alloc((size_t)BN * C * 2);
  _Float16* k16 = (_Float16*)alloc((size_t)BN * C * 2);
  _Float16* v16 = (_Float16*)alloc((size_t)BN * C * 2);
  _Float16* Wqt = (_Float16*)alloc((size_t)C * C * 2);
  _Float16* Wkt = (_Float16*)alloc((size_t)C * C * 2);
  _Float16* Wvt = (_Float16*)alloc((size_t)C * C * 2);
  _Float16* Wpt = (_Float16*)alloc((size_t)C * C * 2);
  _Float16* W1t = (_Float16*)alloc((size_t)C * DFF * 2);
  _Float16* W2t = (_Float16*)alloc((size_t)DFF * C * 2);
  _Float16* Q16 = (_Float16*)alloc((size_t)BN * C * 2);
  _Float16* K16 = (_Float16*)alloc((size_t)BN * C * 2);
  _Float16* V16 = (_Float16*)alloc((size_t)BN * C * 2);
  _Float16* Vt16 = (_Float16*)alloc((size_t)BN * C * 2);
  _Float16* attn16 = (_Float16*)alloc((size_t)BN * C * 2);
  _Float16* h16 = (_Float16*)alloc((size_t)BN * C * 2);
  _Float16* mid16 = (_Float16*)alloc((size_t)BN * DFF * 2);
  float* xres = (float*)alloc((size_t)BN * C * 4);

  layernorm_f16_kernel<<<BN, 256, 0, stream>>>(q, ln1_w, ln1_b, qn16, C);

  cvt_f16_kernel<<<4096, 256, 0, stream>>>(k, k16, (size_t)BN * C);
  cvt_f16_kernel<<<4096, 256, 0, stream>>>(v, v16, (size_t)BN * C);

  cvt_transpose_kernel<<<1024, 256, 0, stream>>>(Wq, Wqt, C, C);
  cvt_transpose_kernel<<<1024, 256, 0, stream>>>(Wk, Wkt, C, C);
  cvt_transpose_kernel<<<1024, 256, 0, stream>>>(Wv, Wvt, C, C);
  cvt_transpose_kernel<<<1024, 256, 0, stream>>>(Wp, Wpt, C, C);
  cvt_transpose_kernel<<<4096, 256, 0, stream>>>(W1, W1t, C, DFF);
  cvt_transpose_kernel<<<4096, 256, 0, stream>>>(W2, W2t, DFF, C);

  dim3 gemm_blk(256);
  dim3 gC(C / 128, BN / 128);
  wmma_gemm_kernel<<<gC, gemm_blk, 0, stream>>>(qn16, Wqt, nullptr, nullptr,
                                                nullptr, Q16, BN, C, C, 0);
  wmma_gemm_kernel<<<gC, gemm_blk, 0, stream>>>(k16, Wkt, nullptr, nullptr,
                                                nullptr, K16, BN, C, C, 0);
  wmma_gemm_kernel<<<gC, gemm_blk, 0, stream>>>(v16, Wvt, nullptr, nullptr,
                                                nullptr, V16, BN, C, C, 0);

  transpose_v_kernel<<<4096, 256, 0, stream>>>(V16, Vt16, B, Mseq, H, D);

  dim3 ga(N / 64, H, B);
  flash_attn_kernel<<<ga, 128, 0, stream>>>(Q16, K16, Vt16, attn16, B, N, Mseq,
                                            H, D);

  wmma_gemm_kernel<<<gC, gemm_blk, 0, stream>>>(attn16, Wpt, bp, q, xres,
                                                nullptr, BN, C, C, 0);

  layernorm_f16_kernel<<<BN, 256, 0, stream>>>(xres, ln2_w, ln2_b, h16, C);

  dim3 gF(DFF / 128, BN / 128);
  wmma_gemm_kernel<<<gF, gemm_blk, 0, stream>>>(h16, W1t, b1, nullptr, nullptr,
                                                mid16, BN, DFF, C, 1);

  wmma_gemm_kernel<<<gC, gemm_blk, 0, stream>>>(mid16, W2t, b2, xres, out,
                                                nullptr, BN, C, DFF, 0);
}
